// Decoder_Layer_87651692577304
// MI455X (gfx1250) — matrix operand order repaired and checked by reading, not compiled, not run on hardware
//
#include <hip/hip_runtime.h>

// ---------------------------------------------------------------------------
// Problem constants (fixed by the reference)
// ---------------------------------------------------------------------------
constexpr int Bc  = 4;
constexpr int Tc  = 2048;
constexpr int Sc  = 2048;
constexpr int Dc  = 1024;
constexpr int Hc  = 16;
constexpr int HDc = 64;
constexpr int DFFc = 4096;
constexpr int Mc  = Bc * Tc;     // 8192 rows (B*T == B*S)
constexpr int BHc = Bc * Hc;     // 64 batch*head

typedef __attribute__((ext_vector_type(16))) __bf16 bf16x16;
typedef __attribute__((ext_vector_type(8)))  float  f32x8;
typedef __attribute__((ext_vector_type(4)))  int    v4i;

union Frag {
  bf16x16 v;
  uint4   u[2];
};

__device__ __forceinline__ f32x8 wmma_bf16(const Frag& a, const Frag& b, f32x8 c) {
  // D = A(16x32 bf16) * B(32x16 bf16) + C(16x16 f32)
  return __builtin_amdgcn_wmma_f32_16x16x32_bf16(
      /*neg_a=*/false, a.v, /*neg_b=*/false, b.v,
      /*c_mod=*/(short)0, c, /*reuse_a=*/false, /*reuse_b=*/false);
}

// ---------------------------------------------------------------------------
// CDNA5 async global->LDS copy (ASYNCcnt-tracked DMA path), with synchronous
// fallback if the toolchain lacks the builtin. 16 bytes per lane per call.
// Builtin signature (from hipcc diagnostic): (v4i addrspace(1)*, v4i addrspace(3)*, imm, imm)
// ---------------------------------------------------------------------------
#if __has_builtin(__builtin_amdgcn_global_load_async_to_lds_b128)
#define HAVE_ASYNC_LDS 1
#else
#define HAVE_ASYNC_LDS 0
#endif

__device__ __forceinline__ void async_copy16(__bf16* lds_dst, const __bf16* gsrc) {
#if HAVE_ASYNC_LDS
  // LDS aperture: generic addr low 32 bits == LDS byte offset (ISA 10.2).
  __builtin_amdgcn_global_load_async_to_lds_b128(
      (__attribute__((address_space(1))) v4i*)(uintptr_t)gsrc,
      (__attribute__((address_space(3))) v4i*)(uint32_t)(uintptr_t)lds_dst,
      0, 0);
#else
  *(uint4*)lds_dst = *(const uint4*)gsrc;
#endif
}

__device__ __forceinline__ void async_wait_all() {
#if HAVE_ASYNC_LDS
#if __has_builtin(__builtin_amdgcn_s_wait_asynccnt)
  __builtin_amdgcn_s_wait_asynccnt(0);
#else
  asm volatile("s_wait_asynccnt 0x0" ::: "memory");
#endif
#endif
}

// ---------------------------------------------------------------------------
// Elementwise: f32 -> bf16
// ---------------------------------------------------------------------------
__global__ void k_cvt(const float* __restrict__ x, __bf16* __restrict__ y, size_t n) {
  size_t i  = (size_t)blockIdx.x * blockDim.x + threadIdx.x;
  size_t st = (size_t)gridDim.x * blockDim.x;
  for (; i < n; i += st) y[i] = (__bf16)x[i];
}

// W[K,N] f32 -> Wt[N,K] bf16  (K-contiguous per output column -> b128 B-frag loads)
__global__ void k_transpose_cvt(const float* __restrict__ w, __bf16* __restrict__ wt,
                                int K, int N) {
  size_t n  = (size_t)K * N;
  size_t i  = (size_t)blockIdx.x * blockDim.x + threadIdx.x;
  size_t st = (size_t)gridDim.x * blockDim.x;
  for (; i < n; i += st) {
    int k  = (int)(i / N);
    int nn = (int)(i % N);
    wt[(size_t)nn * K + k] = (__bf16)w[i];
  }
}

// ---------------------------------------------------------------------------
// GEMM:  C[M,N] = A_bf16[M,K] @ Wt_bf16[N,K]^T + bias  (+optional ReLU)
// 256 threads (8 waves), 128x128 tile, K-step 32, double-buffered LDS with
// async global->LDS prefetch of tile k+1 overlapped with WMMAs on tile k.
// Wave (wr in 0..3, wc in 0..1) computes rows wr*32..+31, cols wc*64..+63.
// ---------------------------------------------------------------------------
template <bool RELU>
__global__ __launch_bounds__(256) void k_gemm(
    const __bf16* __restrict__ A, const __bf16* __restrict__ Bt,
    const float* __restrict__ bias,
    float* __restrict__ Cf, __bf16* __restrict__ Cb,
    int Mdim, int Ndim, int Kdim) {
  __shared__ __bf16 As[2][128 * 40];   // 128 rows x 32 k (+8 pad), ping-pong
  __shared__ __bf16 Bs[2][128 * 40];   // 128 cols x 32 k (+8 pad), ping-pong

  const int tid  = threadIdx.x;
  const int lane = tid & 31;
  const int wave = tid >> 5;
  const int wr   = wave >> 1;
  const int wc   = wave & 1;
  const int half = lane >> 4;   // 0|1 (half-wave)
  const int l16  = lane & 15;
  const int mBase = blockIdx.y * 128;
  const int nBase = blockIdx.x * 128;

  auto stage = [&](int buf, int kk) {
#pragma unroll
    for (int p = 0; p < 2; ++p) {
      int idx = p * 256 + tid;          // 0..511
      int row = idx >> 2;
      int kg  = (idx & 3) * 8;
      async_copy16(&As[buf][row * 40 + kg],
                   &A[(size_t)(mBase + row) * Kdim + kk + kg]);
      async_copy16(&Bs[buf][row * 40 + kg],
                   &Bt[(size_t)(nBase + row) * Kdim + kk + kg]);
    }
  };

  f32x8 acc[2][4] = {};

  stage(0, 0);
  async_wait_all();
  __syncthreads();

  int cur = 0;
  for (int kk = 0; kk < Kdim; kk += 32) {
    if (kk + 32 < Kdim) stage(cur ^ 1, kk + 32);   // prefetch next tile

    Frag a[2], b[4];
#pragma unroll
    for (int i = 0; i < 2; ++i) {
      int r  = wr * 32 + i * 16 + l16;
      int kb = half * 16;                // lanes<16: K 0..15 ; >=16: 16..31
      a[i].u[0] = *(const uint4*)&As[cur][r * 40 + kb];
      a[i].u[1] = *(const uint4*)&As[cur][r * 40 + kb + 8];
    }
#pragma unroll
    for (int j = 0; j < 4; ++j) {
      int n  = wc * 64 + j * 16 + l16;
      int kb = half * 16;                // lanes<16: K 0..15 ; >=16: 16..31
      b[j].u[0] = *(const uint4*)&Bs[cur][n * 40 + kb];
      b[j].u[1] = *(const uint4*)&Bs[cur][n * 40 + kb + 8];
    }
#pragma unroll
    for (int i = 0; i < 2; ++i)
#pragma unroll
      for (int j = 0; j < 4; ++j)
        acc[i][j] = wmma_bf16(a[i], b[j], acc[i][j]);

    async_wait_all();
    __syncthreads();
    cur ^= 1;
  }

  // epilogue: C layout — lanes<16: M=m0+r, N=l16 ; lanes>=16: M=m0+8+r
  const int mOff = half * 8;
#pragma unroll
  for (int i = 0; i < 2; ++i) {
#pragma unroll
    for (int j = 0; j < 4; ++j) {
      int gn   = nBase + wc * 64 + j * 16 + l16;
      float bv = bias ? bias[gn] : 0.f;
#pragma unroll
      for (int r = 0; r < 8; ++r) {
        int gm  = mBase + wr * 32 + i * 16 + mOff + r;
        float v = acc[i][j][r] + bv;
        if (RELU) v = v > 0.f ? v : 0.f;
        size_t o = (size_t)gm * Ndim + gn;
        if (Cf) Cf[o] = v;
        if (Cb) Cb[o] = (__bf16)v;
      }
    }
  }
}

// ---------------------------------------------------------------------------
// Flash attention.  Q,K bf16 [BH, Tq/Tk, 64]; Vt bf16 [BH, 64, Tk]; O f32.
// 128 threads = 4 waves; each wave owns 16 q-rows; 64-key blocks staged with
// async global->LDS copies.
// ---------------------------------------------------------------------------
__global__ __launch_bounds__(128) void k_flash(
    const __bf16* __restrict__ Q, const __bf16* __restrict__ K,
    const __bf16* __restrict__ Vt, float* __restrict__ O,
    int Tq, int Tk, float scale) {
  __shared__ __bf16 Ks[64 * 72];        // [key][hd]
  __shared__ __bf16 Vs[64 * 72];        // [hd][key]
  __shared__ __bf16 Ps[4][16 * 72];     // per-wave P staging [row][key]

  const int tid  = threadIdx.x;
  const int lane = tid & 31;
  const int wave = tid >> 5;
  const int half = lane >> 4;
  const int l16  = lane & 15;
  const int bh     = blockIdx.y;
  const int qBlock = blockIdx.x * 64;

  const __bf16* Qb  = Q  + (size_t)bh * Tq * 64;
  const __bf16* Kb  = K  + (size_t)bh * Tk * 64;
  const __bf16* Vtb = Vt + (size_t)bh * 64 * Tk;
  float*        Ob  = O  + (size_t)bh * Tq * 64;

  // preload this wave's Q fragments (16 rows x 64, two k=32 chunks)
  Frag qf[2];
  {
    int r = qBlock + wave * 16 + l16;
#pragma unroll
    for (int c = 0; c < 2; ++c) {
      int kb = c * 32 + half * 16;
      qf[c].u[0] = *(const uint4*)&Qb[(size_t)r * 64 + kb];
      qf[c].u[1] = *(const uint4*)&Qb[(size_t)r * 64 + kb + 8];
    }
  }

  f32x8 o[4] = {};
  float mrow[8], lrow[8];
#pragma unroll
  for (int r = 0; r < 8; ++r) { mrow[r] = -1e30f; lrow[r] = 0.f; }

  for (int kb0 = 0; kb0 < Tk; kb0 += 64) {
    // stage K block [64 keys x 64 hd] and Vt block [64 hd x 64 keys]
#pragma unroll
    for (int p = 0; p < 4; ++p) {
      int idx = p * 128 + tid;           // 0..511
      int row = idx >> 3;
      int sg  = (idx & 7) * 8;
      async_copy16(&Ks[row * 72 + sg], &Kb[(size_t)(kb0 + row) * 64 + sg]);
      async_copy16(&Vs[row * 72 + sg], &Vtb[(size_t)row * Tk + kb0 + sg]);
    }
    async_wait_all();
    __syncthreads();

    // S = Q * K^T   (B column n == K row -> contiguous)
    f32x8 s[4];
#pragma unroll
    for (int j = 0; j < 4; ++j) {
      int key = j * 16 + l16;
      Frag b0, b1;
      b0.u[0] = *(const uint4*)&Ks[key * 72 + half * 16];
      b0.u[1] = *(const uint4*)&Ks[key * 72 + half * 16 + 8];
      b1.u[0] = *(const uint4*)&Ks[key * 72 + 32 + half * 16];
      b1.u[1] = *(const uint4*)&Ks[key * 72 + 32 + half * 16 + 8];
      f32x8 z = {};
      z = wmma_bf16(qf[0], b0, z);
      z = wmma_bf16(qf[1], b1, z);
      s[j] = z;
    }

    // online softmax per owned row (width-16 shuffles stay within half-wave)
    float alpha[8];
#pragma unroll
    for (int r = 0; r < 8; ++r) {
      float mloc = -1e30f;
#pragma unroll
      for (int j = 0; j < 4; ++j) { s[j][r] *= scale; mloc = fmaxf(mloc, s[j][r]); }
#pragma unroll
      for (int d1 = 1; d1 < 16; d1 <<= 1) mloc = fmaxf(mloc, __shfl_xor(mloc, d1, 16));
      float mnew = fmaxf(mrow[r], mloc);
      alpha[r] = __expf(mrow[r] - mnew);
      mrow[r]  = mnew;
      float rs = 0.f;
#pragma unroll
      for (int j = 0; j < 4; ++j) { float p = __expf(s[j][r] - mnew); s[j][r] = p; rs += p; }
#pragma unroll
      for (int d1 = 1; d1 < 16; d1 <<= 1) rs += __shfl_xor(rs, d1, 16);
      lrow[r] = lrow[r] * alpha[r] + rs;
    }
#pragma unroll
    for (int n = 0; n < 4; ++n)
#pragma unroll
      for (int r = 0; r < 8; ++r) o[n][r] *= alpha[r];

    // re-layout P (C-layout -> A-fragment layout) through per-wave LDS tile
#pragma unroll
    for (int j = 0; j < 4; ++j)
#pragma unroll
      for (int r = 0; r < 8; ++r)
        Ps[wave][(half * 8 + r) * 72 + j * 16 + l16] = (__bf16)s[j][r];
    __builtin_amdgcn_wave_barrier();   // wave-synchronous LDS RAW (in-order per wave)

    Frag pa[2];
#pragma unroll
    for (int c = 0; c < 2; ++c) {
      int kb = c * 32 + half * 16;
      pa[c].u[0] = *(const uint4*)&Ps[wave][l16 * 72 + kb];
      pa[c].u[1] = *(const uint4*)&Ps[wave][l16 * 72 + kb + 8];
    }
    // O += P * V   (B column n == Vt row -> contiguous)
#pragma unroll
    for (int n = 0; n < 4; ++n) {
      int hr = n * 16 + l16;
      Frag b0, b1;
      b0.u[0] = *(const uint4*)&Vs[hr * 72 + half * 16];
      b0.u[1] = *(const uint4*)&Vs[hr * 72 + half * 16 + 8];
      b1.u[0] = *(const uint4*)&Vs[hr * 72 + 32 + half * 16];
      b1.u[1] = *(const uint4*)&Vs[hr * 72 + 32 + half * 16 + 8];
      o[n] = wmma_bf16(pa[0], b0, o[n]);
      o[n] = wmma_bf16(pa[1], b1, o[n]);
    }
    __syncthreads();
  }

  const int mOff = half * 8;
#pragma unroll
  for (int r = 0; r < 8; ++r) {
    float linv = 1.f / lrow[r];
    int row = qBlock + wave * 16 + mOff + r;
#pragma unroll
    for (int n = 0; n < 4; ++n)
      Ob[(size_t)row * 64 + n * 16 + l16] = o[n][r] * linv;
  }
}

// ---------------------------------------------------------------------------
// Head split / merge helpers
// ---------------------------------------------------------------------------
__global__ void k_split_qkv(const __bf16* __restrict__ qkv,
                            __bf16* __restrict__ Q, __bf16* __restrict__ K,
                            __bf16* __restrict__ Vt) {
  size_t n = (size_t)BHc * Tc * HDc;
  size_t i = (size_t)blockIdx.x * 256 + threadIdx.x;
  if (i >= n) return;
  int d  = (int)(i & 63);
  size_t r6 = i >> 6;
  int t  = (int)(r6 & (Tc - 1));
  int bh = (int)(r6 >> 11);
  int h = bh & (Hc - 1), b = bh >> 4;
  size_t src = (size_t)(b * Tc + t) * (3 * Dc) + h * (3 * HDc) + d;
  Q[i] = qkv[src];
  K[i] = qkv[src + HDc];
  __bf16 vv = qkv[src + 2 * HDc];
  Vt[((size_t)bh * HDc + d) * Tc + t] = vv;
}

__global__ void k_split_kv(const __bf16* __restrict__ kv,
                           __bf16* __restrict__ Kc, __bf16* __restrict__ Vct) {
  size_t n = (size_t)BHc * Sc * HDc;
  size_t i = (size_t)blockIdx.x * 256 + threadIdx.x;
  if (i >= n) return;
  int d  = (int)(i & 63);
  size_t r6 = i >> 6;
  int t  = (int)(r6 & (Sc - 1));
  int bh = (int)(r6 >> 11);
  int h = bh & (Hc - 1), b = bh >> 4;
  size_t src = (size_t)(b * Sc + t) * (2 * Dc) + h * (2 * HDc) + d;
  Kc[i] = kv[src];
  Vct[((size_t)bh * HDc + d) * Sc + t] = kv[src + HDc];
}

__global__ void k_split_q(const __bf16* __restrict__ qc, __bf16* __restrict__ Qc) {
  size_t n = (size_t)BHc * Tc * HDc;
  size_t i = (size_t)blockIdx.x * 256 + threadIdx.x;
  if (i >= n) return;
  int d  = (int)(i & 63);
  size_t r6 = i >> 6;
  int t  = (int)(r6 & (Tc - 1));
  int bh = (int)(r6 >> 11);
  int h = bh & (Hc - 1), b = bh >> 4;
  Qc[i] = qc[(size_t)(b * Tc + t) * Dc + h * HDc + d];
}

// O f32 [B,H,T,64] -> bf16 [B,T,D]
__global__ void k_merge(const float* __restrict__ Oin, __bf16* __restrict__ outb) {
  size_t n = (size_t)Mc * Dc;
  size_t i = (size_t)blockIdx.x * 256 + threadIdx.x;
  if (i >= n) return;
  int d = (int)(i & 63);
  int h = (int)((i >> 6) & (Hc - 1));
  size_t bt = i >> 10;
  int b = (int)(bt >> 11);
  int t = (int)(bt & (Tc - 1));
  outb[i] = (__bf16)Oin[(((size_t)(b * Hc + h)) * Tc + t) * HDc + d];
}

// ---------------------------------------------------------------------------
// Fused residual add + LayerNorm. One block per row of D=1024.
// ---------------------------------------------------------------------------
__global__ __launch_bounds__(256) void k_add_ln(
    const float* __restrict__ a, const float* __restrict__ b,
    const float* __restrict__ gamma, const float* __restrict__ beta,
    float* __restrict__ outf, __bf16* __restrict__ outb) {
  const int row  = blockIdx.x;
  const int lane = threadIdx.x & 31;
  const int wave = threadIdx.x >> 5;
  const float* ar = a + (size_t)row * Dc;
  const float* br = b + (size_t)row * Dc;

  float s0 = 0.f, s1 = 0.f;
  for (int i = threadIdx.x; i < Dc; i += 256) {
    float v = ar[i] + br[i];
    s0 += v; s1 += v * v;
  }
  for (int d1 = 1; d1 < 32; d1 <<= 1) {
    s0 += __shfl_xor(s0, d1, 32);
    s1 += __shfl_xor(s1, d1, 32);
  }
  __shared__ float w0[8], w1[8];
  if (lane == 0) { w0[wave] = s0; w1[wave] = s1; }
  __syncthreads();
  float t0 = 0.f, t1 = 0.f;
#pragma unroll
  for (int i = 0; i < 8; ++i) { t0 += w0[i]; t1 += w1[i]; }
  const float mean = t0 / Dc;
  const float var  = t1 / Dc - mean * mean;
  const float inv  = rsqrtf(var + 1e-5f);

  for (int i = threadIdx.x; i < Dc; i += 256) {
    float v = ar[i] + br[i];
    float y = gamma[i] * (v - mean) * inv + beta[i];
    size_t o = (size_t)row * Dc + i;
    outf[o] = y;
    if (outb) outb[o] = (__bf16)y;
  }
}

// ---------------------------------------------------------------------------
// Host launch
// ---------------------------------------------------------------------------
extern "C" void kernel_launch(void* const* d_in, const int* in_sizes, int n_in,
                              void* d_out, int out_size, void* d_ws, size_t ws_size,
                              hipStream_t stream) {
  const float* x_enc  = (const float*)d_in[0];
  const float* x_dec  = (const float*)d_in[1];
  const float* qkv_w  = (const float*)d_in[2];
  const float* qkv_bb = (const float*)d_in[3];
  const float* sa_o_w = (const float*)d_in[4];
  const float* sa_o_b = (const float*)d_in[5];
  const float* g1     = (const float*)d_in[6];
  const float* be1    = (const float*)d_in[7];
  const float* q_w    = (const float*)d_in[8];
  const float* q_b    = (const float*)d_in[9];
  const float* kv_w   = (const float*)d_in[10];
  const float* kv_bb  = (const float*)d_in[11];
  const float* ca_o_w = (const float*)d_in[12];
  const float* ca_o_b = (const float*)d_in[13];
  const float* g2     = (const float*)d_in[14];
  const float* be2    = (const float*)d_in[15];
  const float* ff_w1  = (const float*)d_in[16];
  const float* ff_b1  = (const float*)d_in[17];
  const float* ff_w2  = (const float*)d_in[18];
  const float* ff_b2  = (const float*)d_in[19];
  const float* g3     = (const float*)d_in[20];
  const float* be3    = (const float*)d_in[21];
  float* out = (float*)d_out;

  char* base = (char*)d_ws;
  size_t off = 0;
  auto alloc = [&](size_t elems, size_t elsz) -> void* {
    void* p = base + off;
    off += ((elems * elsz + 255) & ~(size_t)255);
    return p;
  };

  // bf16 transposed weights
  __bf16* wt_qkv = (__bf16*)alloc((size_t)Dc * 3 * Dc, 2);
  __bf16* wt_sao = (__bf16*)alloc((size_t)Dc * Dc, 2);
  __bf16* wt_q   = (__bf16*)alloc((size_t)Dc * Dc, 2);
  __bf16* wt_kv  = (__bf16*)alloc((size_t)Dc * 2 * Dc, 2);
  __bf16* wt_cao = (__bf16*)alloc((size_t)Dc * Dc, 2);
  __bf16* wt_ff1 = (__bf16*)alloc((size_t)Dc * DFFc, 2);
  __bf16* wt_ff2 = (__bf16*)alloc((size_t)DFFc * Dc, 2);
  // activations
  __bf16* xdec_b  = (__bf16*)alloc((size_t)Mc * Dc, 2);
  __bf16* xenc_b  = (__bf16*)alloc((size_t)Mc * Dc, 2);
  __bf16* qkv_bf  = (__bf16*)alloc((size_t)Mc * 3 * Dc, 2);
  __bf16* Qh      = (__bf16*)alloc((size_t)BHc * Tc * HDc, 2);
  __bf16* Kh      = (__bf16*)alloc((size_t)BHc * Tc * HDc, 2);
  __bf16* Vth     = (__bf16*)alloc((size_t)BHc * HDc * Tc, 2);
  float*  attnO   = (float*) alloc((size_t)BHc * Tc * HDc, 4);
  __bf16* merged  = (__bf16*)alloc((size_t)Mc * Dc, 2);
  float*  tmpf    = (float*) alloc((size_t)Mc * Dc, 4);
  float*  res1    = (float*) alloc((size_t)Mc * Dc, 4);
  __bf16* res1_b  = (__bf16*)alloc((size_t)Mc * Dc, 2);
  __bf16* kv_bf   = (__bf16*)alloc((size_t)Mc * 2 * Dc, 2);
  __bf16* Kch     = (__bf16*)alloc((size_t)BHc * Sc * HDc, 2);
  __bf16* Vcth    = (__bf16*)alloc((size_t)BHc * HDc * Sc, 2);
  __bf16* qc_b    = (__bf16*)alloc((size_t)Mc * Dc, 2);
  __bf16* Qch     = (__bf16*)alloc((size_t)BHc * Tc * HDc, 2);
  float*  res2    = (float*) alloc((size_t)Mc * Dc, 4);
  __bf16* res2_b  = (__bf16*)alloc((size_t)Mc * Dc, 2);
  __bf16* ff1_b   = (__bf16*)alloc((size_t)Mc * DFFc, 2);

  auto NB = [](size_t n) { return dim3((unsigned)((n + 255) / 256)); };
  const size_t nMD  = (size_t)Mc * Dc;
  const size_t nHTD = (size_t)BHc * Tc * HDc;

  // 0) precision conversion + weight transpose
  k_transpose_cvt<<<NB((size_t)Dc * 3 * Dc), 256, 0, stream>>>(qkv_w,  wt_qkv, Dc, 3 * Dc);
  k_transpose_cvt<<<NB((size_t)Dc * Dc),     256, 0, stream>>>(sa_o_w, wt_sao, Dc, Dc);
  k_transpose_cvt<<<NB((size_t)Dc * Dc),     256, 0, stream>>>(q_w,    wt_q,   Dc, Dc);
  k_transpose_cvt<<<NB((size_t)Dc * 2 * Dc), 256, 0, stream>>>(kv_w,   wt_kv,  Dc, 2 * Dc);
  k_transpose_cvt<<<NB((size_t)Dc * Dc),     256, 0, stream>>>(ca_o_w, wt_cao, Dc, Dc);
  k_transpose_cvt<<<NB((size_t)Dc * DFFc),   256, 0, stream>>>(ff_w1,  wt_ff1, Dc, DFFc);
  k_transpose_cvt<<<NB((size_t)DFFc * Dc),   256, 0, stream>>>(ff_w2,  wt_ff2, DFFc, Dc);
  k_cvt<<<NB(nMD), 256, 0, stream>>>(x_dec, xdec_b, nMD);
  k_cvt<<<NB(nMD), 256, 0, stream>>>(x_enc, xenc_b, nMD);

  const float scale = 0.125f;  // 1/sqrt(64)

  // 1) self attention
  k_gemm<false><<<dim3(3 * Dc / 128, Mc / 128), 256, 0, stream>>>(
      xdec_b, wt_qkv, qkv_bb, nullptr, qkv_bf, Mc, 3 * Dc, Dc);
  k_split_qkv<<<NB(nHTD), 256, 0, stream>>>(qkv_bf, Qh, Kh, Vth);
  k_flash<<<dim3(Tc / 64, BHc), 128, 0, stream>>>(Qh, Kh, Vth, attnO, Tc, Tc, scale);
  k_merge<<<NB(nMD), 256, 0, stream>>>(attnO, merged);
  k_gemm<false><<<dim3(Dc / 128, Mc / 128), 256, 0, stream>>>(
      merged, wt_sao, sa_o_b, tmpf, nullptr, Mc, Dc, Dc);
  k_add_ln<<<Mc, 256, 0, stream>>>(tmpf, x_dec, g1, be1, res1, res1_b);

  // 2) cross attention
  k_gemm<false><<<dim3(2 * Dc / 128, Mc / 128), 256, 0, stream>>>(
      xenc_b, wt_kv, kv_bb, nullptr, kv_bf, Mc, 2 * Dc, Dc);
  k_split_kv<<<NB((size_t)BHc * Sc * HDc), 256, 0, stream>>>(kv_bf, Kch, Vcth);
  k_gemm<false><<<dim3(Dc / 128, Mc / 128), 256, 0, stream>>>(
      res1_b, wt_q, q_b, nullptr, qc_b, Mc, Dc, Dc);
  k_split_q<<<NB(nHTD), 256, 0, stream>>>(qc_b, Qch);
  k_flash<<<dim3(Tc / 64, BHc), 128, 0, stream>>>(Qch, Kch, Vcth, attnO, Tc, Sc, scale);
  k_merge<<<NB(nMD), 256, 0, stream>>>(attnO, merged);
  k_gemm<false><<<dim3(Dc / 128, Mc / 128), 256, 0, stream>>>(
      merged, wt_cao, ca_o_b, tmpf, nullptr, Mc, Dc, Dc);
  k_add_ln<<<Mc, 256, 0, stream>>>(tmpf, res1, g2, be2, res2, res2_b);

  // 3) feed forward
  k_gemm<true><<<dim3(DFFc / 128, Mc / 128), 256, 0, stream>>>(
      res2_b, wt_ff1, ff_b1, nullptr, ff1_b, Mc, DFFc, Dc);
  k_gemm<false><<<dim3(Dc / 128, Mc / 128), 256, 0, stream>>>(
      ff1_b, wt_ff2, ff_b2, tmpf, nullptr, Mc, Dc, DFFc);
  k_add_ln<<<Mc, 256, 0, stream>>>(tmpf, res2, g3, be3, out, nullptr);
}
